// SphericalHarmonicsAttentionBias_41764261986702
// MI455X (gfx1250) — matrix operand order repaired and checked by reading, not compiled, not run on hardware
//
#include <hip/hip_runtime.h>
#include <math.h>

typedef __attribute__((ext_vector_type(16))) _Float16 v16h;
typedef __attribute__((ext_vector_type(8)))  float    v8f;
typedef __attribute__((ext_vector_type(4)))  float    v4f;

#define HD 64   // hidden dim (fixed by reference)

// e3nn 'integral' normalization constants
#define SH_C0 0.28209479177387814f          // 0.5/sqrt(pi)
#define SH_C1 0.4886025119029199f           // sqrt(3/(4pi))
#define SH_C2 0.6307831305050401f           // sqrt(5/(4pi))
#define SH_S3 1.7320508075688772f           // sqrt(3)
#define SH_EPS 1e-6f

__device__ __forceinline__ float silu_f(float x) {
    // x * sigmoid(x) with native v_rcp_f32 / v_exp_f32 (no IEEE div expansion)
    return x * __builtin_amdgcn_rcpf(1.0f + __expf(-x));
}

__launch_bounds__(256)
__global__ void sh_attn_bias_kernel(const float* __restrict__ coords,
                                    const float* __restrict__ w1,
                                    const float* __restrict__ b1,
                                    const float* __restrict__ w2,
                                    const float* __restrict__ b2,
                                    const float* __restrict__ w3,
                                    const float* __restrict__ b3,
                                    float* __restrict__ out,
                                    int S, int H,
                                    unsigned int tiles,   // B*S*(S/16)
                                    unsigned int tpr)     // S/16 tiles per row
{
    const int lane = threadIdx.x & 31;
    const int wave = threadIdx.x >> 5;
    const int hi   = lane >> 4;     // which 16-lane half
    const int lm   = lane & 15;

    // per-wave transpose staging: 16 rows x 64 cols f16, padded to 68 halves/row
    __shared__ _Float16 hbuf[8][16][68];

    // ---------------- weight fragments (B-layout: K = e + 16*hi, N = lm) ----------
    v16h wf1[4];
    #pragma unroll
    for (int nt = 0; nt < 4; ++nt) {
        #pragma unroll
        for (int e = 0; e < 16; ++e) {
            int k = e + hi * 16;
            float v = (k < 9) ? w1[k * HD + nt * 16 + lm] : 0.0f;
            wf1[nt][e] = (_Float16)v;
        }
    }
    v16h wf2[2][4];
    #pragma unroll
    for (int kc = 0; kc < 2; ++kc)
        #pragma unroll
        for (int nt = 0; nt < 4; ++nt)
            #pragma unroll
            for (int e = 0; e < 16; ++e) {
                int k = kc * 32 + e + hi * 16;
                wf2[kc][nt][e] = (_Float16)w2[k * HD + nt * 16 + lm];
            }
    v16h wf3[2];
    #pragma unroll
    for (int kc = 0; kc < 2; ++kc)
        #pragma unroll
        for (int e = 0; e < 16; ++e) {
            int k = kc * 32 + e + hi * 16;
            float v = (lm < H) ? w3[k * H + lm] : 0.0f;
            wf3[kc][e] = (_Float16)v;
        }
    float bb1[4], bb2[4];
    #pragma unroll
    for (int nt = 0; nt < 4; ++nt) {
        bb1[nt] = b1[nt * 16 + lm];
        bb2[nt] = b2[nt * 16 + lm];
    }
    const float bb3 = (lm < H) ? b3[lm] : 0.0f;

    const unsigned int wstride = gridDim.x * (blockDim.x >> 5);

    for (unsigned int t = blockIdx.x * (blockDim.x >> 5) + wave;
         t < tiles; t += wstride) {
        // tile -> (b, i, j0) with 32-bit uniform math:
        //   t = (b*S + i)*tpr + jt ,  j0 = 16*jt
        unsigned int jt   = t % tpr;
        unsigned int rest = t / tpr;
        unsigned int i0   = rest % (unsigned int)S;
        unsigned int b0   = rest / (unsigned int)S;
        int j0 = (int)(jt << 4);

        // ---------- spherical harmonics for this lane's row (j = j0 + lm) ---------
        const float* ci = coords + ((long)b0 * S + i0) * 3;          // wave-uniform
        const float* cj = coords + ((long)b0 * S + j0 + lm) * 3;     // per-lane
        float rx = ci[0] - cj[0];
        float ry = ci[1] - cj[1];
        float rz = ci[2] - cj[2];
        float nrm = __builtin_amdgcn_sqrtf(rx * rx + ry * ry + rz * rz);
        float inv = __builtin_amdgcn_rcpf(nrm + SH_EPS);
        float ux = rx * inv, uy = ry * inv, uz = rz * inv;
        float un = __builtin_amdgcn_sqrtf(ux * ux + uy * uy + uz * uz);
        float inv2 = __builtin_amdgcn_rcpf(fmaxf(un, SH_EPS));
        ux *= inv2; uy *= inv2; uz *= inv2;

        float sh0 = SH_C0;
        float sh1 = SH_C1 * ux;
        float sh2 = SH_C1 * uy;
        float sh3 = SH_C1 * uz;
        float sh4 = SH_C2 * SH_S3 * ux * uz;
        float sh5 = SH_C2 * SH_S3 * ux * uy;
        float sh6 = SH_C2 * (uy * uy - 0.5f * (ux * ux + uz * uz));
        float sh7 = SH_C2 * SH_S3 * uy * uz;
        float sh8 = SH_C2 * (SH_S3 * 0.5f) * (uz * uz - ux * ux);

        // ---------- build A1 fragment (16x32 f16, K padded 9->32) -----------------
        v16h a1;
        #pragma unroll
        for (int e = 0; e < 16; ++e) a1[e] = (_Float16)0.0f;
        if (hi == 0) {
            a1[0] = (_Float16)sh0; a1[1] = (_Float16)sh1;
            a1[2] = (_Float16)sh2; a1[3] = (_Float16)sh3;
            a1[4] = (_Float16)sh4; a1[5] = (_Float16)sh5;
            a1[6] = (_Float16)sh6; a1[7] = (_Float16)sh7;
            a1[8] = (_Float16)sh8;   // K=8
        }

        // ---------- layer 1: [16x9] x [9x64] -> silu -> hbuf ----------------------
        #pragma unroll
        for (int nt = 0; nt < 4; ++nt) {
            v8f z = {};
            v8f c = __builtin_amdgcn_wmma_f32_16x16x32_f16(
                        false, a1, false, wf1[nt], (short)0, z, false, false);
            #pragma unroll
            for (int v = 0; v < 8; ++v) {
                float x = c[v] + bb1[nt];
                hbuf[wave][v + hi * 8][nt * 16 + lm] = (_Float16)silu_f(x);
            }
        }

        // ---------- layer 2: [16x64] x [64x64] -> silu -> hbuf --------------------
        v16h a2[2];
        #pragma unroll
        for (int kc = 0; kc < 2; ++kc)
            #pragma unroll
            for (int e = 0; e < 16; ++e) {
                int k = kc * 32 + e + hi * 16;
                a2[kc][e] = hbuf[wave][lm][k];
            }
        #pragma unroll
        for (int nt = 0; nt < 4; ++nt) {
            v8f z = {};
            v8f c = __builtin_amdgcn_wmma_f32_16x16x32_f16(
                        false, a2[0], false, wf2[0][nt], (short)0, z, false, false);
            c = __builtin_amdgcn_wmma_f32_16x16x32_f16(
                        false, a2[1], false, wf2[1][nt], (short)0, c, false, false);
            #pragma unroll
            for (int v = 0; v < 8; ++v) {
                float x = c[v] + bb2[nt];
                hbuf[wave][v + hi * 8][nt * 16 + lm] = (_Float16)silu_f(x);
            }
        }

        // ---------- layer 3: [16x64] x [64x8(pad16)] ------------------------------
        v16h a3[2];
        #pragma unroll
        for (int kc = 0; kc < 2; ++kc)
            #pragma unroll
            for (int e = 0; e < 16; ++e) {
                int k = kc * 32 + e + hi * 16;
                a3[kc][e] = hbuf[wave][lm][k];
            }
        v8f z3 = {};
        v8f c3 = __builtin_amdgcn_wmma_f32_16x16x32_f16(
                     false, a3[0], false, wf3[0], (short)0, z3, false, false);
        c3 = __builtin_amdgcn_wmma_f32_16x16x32_f16(
                     false, a3[1], false, wf3[1], (short)0, c3, false, false);

        // ---------- scatter to out[b][head][i][j0 + v + 8*hi] ---------------------
        if (lm < H) {
            float* o = out + (((long)b0 * H + lm) * S + i0) * (long)S + j0 + hi * 8;
            v4f o0 = { c3[0] + bb3, c3[1] + bb3, c3[2] + bb3, c3[3] + bb3 };
            v4f o1 = { c3[4] + bb3, c3[5] + bb3, c3[6] + bb3, c3[7] + bb3 };
            *(v4f*)(o)     = o0;
            *(v4f*)(o + 4) = o1;
        }
    }
}

extern "C" void kernel_launch(void* const* d_in, const int* in_sizes, int n_in,
                              void* d_out, int out_size, void* d_ws, size_t ws_size,
                              hipStream_t stream) {
    const float* coords = (const float*)d_in[0];
    const float* w1 = (const float*)d_in[1];
    const float* b1 = (const float*)d_in[2];
    const float* w2 = (const float*)d_in[3];
    const float* b2 = (const float*)d_in[4];
    const float* w3 = (const float*)d_in[5];
    const float* b3 = (const float*)d_in[6];
    float* out = (float*)d_out;

    // derive shapes: coords[B*S*3], w1[9*Hd], b1[Hd], w2[Hd*Hd], b2[Hd],
    // w3[Hd*H], b3[H]; out = B*H*S*S
    const int  H  = in_sizes[6];
    const long BS = (long)in_sizes[0] / 3;          // B*S
    const long S  = (long)out_size / (BS * (long)H);
    const int  B  = (int)(BS / S);

    const unsigned int tpr   = (unsigned int)(S / 16);       // tiles per row
    const unsigned int tiles = (unsigned int)((long)B * S * tpr);

    const int blocks = 2048;   // 8 waves/block, grid-stride over tiles
    sh_attn_bias_kernel<<<blocks, 256, 0, stream>>>(
        coords, w1, b1, w2, b2, w3, b3, out, (int)S, H, tiles, tpr);
}
